// Self_Attn_Spatial_11115375362943
// MI455X (gfx1250) — matrix operand order repaired and checked by reading, not compiled, not run on hardware
//
#include <hip/hip_runtime.h>

typedef __attribute__((ext_vector_type(16))) _Float16 v16h;
typedef __attribute__((ext_vector_type(8)))  _Float16 v8h;
typedef __attribute__((ext_vector_type(8)))  float    v8f;
typedef __attribute__((ext_vector_type(4)))  float    fvec4;

#define B_   8
#define C_   256
#define L_   2048
#define O_   32
#define M320 320

// A-operand (16-bit, 16x32) element->K mapping (matches B operand loads):
// v16h elements 0..15 hold K = 16*hf .. 16*hf+15 (contiguous)
__device__ __forceinline__ int kprime(int e, int hf) {
  return e + 16 * hf;
}

// ---------------------------------------------------------------------------
// Pack [wq;wk;wv] (320x256 f32) into f16, pre-swizzled into WMMA A layout.
// ---------------------------------------------------------------------------
__global__ void prepack_w(const float* __restrict__ wq, const float* __restrict__ wk,
                          const float* __restrict__ wv, _Float16* __restrict__ Wh) {
  int t = blockIdx.x * blockDim.x + threadIdx.x;          // 81920 threads
  int e = t & 15, lane = (t >> 4) & 31, kt = (t >> 9) & 7, mt = t >> 12;
  int m = mt * 16 + (lane & 15);
  int c = kt * 32 + kprime(e, lane >> 4);
  float v;
  if (m < 32)      v = wq[m * C_ + c];
  else if (m < 64) v = wk[(m - 32) * C_ + c];
  else             v = wv[(m - 64) * C_ + c];
  Wh[t] = (_Float16)v;
}

// ---------------------------------------------------------------------------
// Projection GEMM: Pall[b][m][l] = sum_c W[m][c]*x[b][c][l] + bias[m] (f16 out)
// Grid: b(8) x mg(2) x lt(8) = 128 blocks, 512 threads = 16 waves.
// Each block owns 160 output rows (10 m-tiles held in registers) so every
// staged x tile is reused by 10 WMMAs per wave -> staging traffic /10.
// ---------------------------------------------------------------------------
__global__ void __launch_bounds__(512) proj_gemm(
    const float* __restrict__ x, const _Float16* __restrict__ Wh,
    const float* __restrict__ bq, const float* __restrict__ bk,
    const float* __restrict__ bv, _Float16* __restrict__ Pall) {
  __shared__ _Float16 xs[256 * 40];   // x tile transposed [l][c], stride 40 -> 16B-aligned b128
  int bx = blockIdx.x;
  int b = bx >> 4, mg = (bx >> 3) & 1, lt = bx & 7;
  int tid = threadIdx.x, wave = tid >> 5, lane = tid & 31;
  int lane15 = lane & 15, hf = lane >> 4;
  v8f acc[10] = {};
  for (int kt = 0; kt < 8; ++kt) {
    // ---- batched, coalesced staging loads: 4x float4 per thread ----
    fvec4 tmp[4];
    #pragma unroll
    for (int it = 0; it < 4; ++it) {                      // 2048 float4 units: 32c x 64(l/4)
      int idx = it * 512 + tid;
      int cc = idx >> 6, l4 = idx & 63;
      tmp[it] = *(const fvec4*)(x + (size_t)(b * C_ + kt * 32 + cc) * L_ + lt * 256 + l4 * 4);
    }
    __syncthreads();                                      // prev tile's reads complete
    #pragma unroll
    for (int it = 0; it < 4; ++it) {
      int idx = it * 512 + tid;
      int cc = idx >> 6, l4 = idx & 63;
      #pragma unroll
      for (int q = 0; q < 4; ++q)
        xs[(l4 * 4 + q) * 40 + cc] = (_Float16)tmp[it][q];
    }
    __syncthreads();
    // ---- one B operand per wave, reused by 10 A tiles ----
    int lloc = wave * 16 + lane15;
    v16h bb = *(const v16h*)(xs + lloc * 40 + hf * 16);
    #pragma unroll
    for (int mt = 0; mt < 10; ++mt) {
      int mtg = mg * 10 + mt;
      v16h a = *(const v16h*)(Wh + ((size_t)(mtg * 8 + kt) * 32 + lane) * 16);
      acc[mt] = __builtin_amdgcn_wmma_f32_16x16x32_f16(false, a, false, bb, (short)0,
                                                       acc[mt], false, false);
    }
  }
  int col = lt * 256 + wave * 16 + lane15;
  #pragma unroll
  for (int mt = 0; mt < 10; ++mt) {
    #pragma unroll
    for (int r = 0; r < 8; ++r) {                         // C/D: VGPR r -> row r + 8*hf
      int m = (mg * 10 + mt) * 16 + 8 * hf + r;
      float bias = (m < 32) ? bq[m] : (m < 64) ? bk[m - 32] : bv[m - 64];
      Pall[(size_t)(b * M320 + m) * L_ + col] = (_Float16)(acc[mt][r] + bias);
    }
  }
}

// ---------------------------------------------------------------------------
// Repack Q rows of Pall into pre-swizzled WMMA A tiles (per 16-row i-tile).
// ---------------------------------------------------------------------------
__global__ void repack_q(const _Float16* __restrict__ Pall, _Float16* __restrict__ Qh) {
  int t = blockIdx.x * blockDim.x + threadIdx.x;          // 524288 threads
  int e = t & 15, lane = (t >> 4) & 31, itl = (t >> 9) & 127, b = t >> 16;
  int o = kprime(e, lane >> 4);
  int i = itl * 16 + (lane & 15);
  Qh[t] = Pall[(size_t)(b * M320 + o) * L_ + i];          // Qh idx == t by construction
}

// K transposed to [b][j][o] so energy B-operand loads are contiguous per lane.
__global__ void repack_k(const _Float16* __restrict__ Pall, _Float16* __restrict__ Kh) {
  int t = blockIdx.x * blockDim.x + threadIdx.x;          // 524288 threads
  int j = t & 2047, o = (t >> 11) & 31, b = t >> 16;
  Kh[((size_t)(b * L_) + j) * 32 + o] = Pall[(size_t)(b * M320 + 32 + o) * L_ + j];
}

// ---------------------------------------------------------------------------
// Fused attention: per (b, 16-row i-tile) workgroup, 16 waves.
// Phase 1: energy row-block [16 x 2048], row-block softmax, exp(P)->f16 LDS.
// Phase 2: out = P x V^T; 1/rowsum and gamma folded into the residual store.
// LDS = exactly 64KB (reduction scratch aliases Pl and is consumed first).
// ---------------------------------------------------------------------------
__global__ void __launch_bounds__(512) attn_fused(
    const float* __restrict__ x, const _Float16* __restrict__ Pall,
    const _Float16* __restrict__ Qh, const _Float16* __restrict__ Kh,
    const float* __restrict__ gamma, float* __restrict__ out) {
  __shared__ _Float16 Pl[16 * 2048];                      // 65536 B
  float* redf = (float*)Pl;                               // [0..255] partials, [256..287] stats
  int b = blockIdx.x >> 7, it = blockIdx.x & 127, i0 = it * 16;
  int tid = threadIdx.x, wave = tid >> 5, lane = tid & 31;
  int lane15 = lane & 15, hf = lane >> 4;

  // ---- Phase 1: S = Q_i * K^T over this wave's 128-j chunk ----
  v16h aQ = *(const v16h*)(Qh + ((size_t)(b * 128 + it) * 32 + lane) * 16);
  v8f S[8];
  #pragma unroll
  for (int st = 0; st < 8; ++st) {
    int jt = wave * 8 + st;
    v16h bK = *(const v16h*)(Kh + ((size_t)(b * L_) + jt * 16 + lane15) * 32 + hf * 16);
    v8f z = {};
    S[st] = __builtin_amdgcn_wmma_f32_16x16x32_f16(false, aQ, false, bK, (short)0, z, false, false);
  }

  // row max: per-lane over 8 subtiles, then across the 16-lane half (j dim)
  float mr[8];
  #pragma unroll
  for (int r = 0; r < 8; ++r) {
    float m = S[0][r];
    #pragma unroll
    for (int st = 1; st < 8; ++st) m = fmaxf(m, S[st][r]);
    #pragma unroll
    for (int d = 1; d < 16; d <<= 1) m = fmaxf(m, __shfl_xor(m, d, 32));
    mr[r] = m;
  }
  if (lane15 == 0) {
    #pragma unroll
    for (int r = 0; r < 8; ++r) redf[wave * 16 + 8 * hf + r] = mr[r];
  }
  __syncthreads();
  if (tid < 16) {
    float m = redf[tid];
    for (int w = 1; w < 16; ++w) m = fmaxf(m, redf[w * 16 + tid]);
    redf[256 + tid] = m;                                  // stats: row max
  }
  __syncthreads();

  // exp + row sums
  float sr[8];
  #pragma unroll
  for (int r = 0; r < 8; ++r) {
    float rm = redf[256 + 8 * hf + r];
    float s = 0.f;
    #pragma unroll
    for (int st = 0; st < 8; ++st) {
      float e = __expf(S[st][r] - rm);
      S[st][r] = e;
      s += e;
    }
    #pragma unroll
    for (int d = 1; d < 16; d <<= 1) s += __shfl_xor(s, d, 32);
    sr[r] = s;
  }
  __syncthreads();                                        // row-max reads done before overwrite
  if (lane15 == 0) {
    #pragma unroll
    for (int r = 0; r < 8; ++r) redf[wave * 16 + 8 * hf + r] = sr[r];
  }
  __syncthreads();
  if (tid < 16) {
    float s = redf[tid];
    for (int w = 1; w < 16; ++w) s += redf[w * 16 + tid];
    redf[272 + tid] = s;                                  // stats: row sum
  }
  __syncthreads();
  float rs[8];                                            // pull sums to regs before Pl overwrite
  #pragma unroll
  for (int r = 0; r < 8; ++r) rs[r] = redf[272 + 8 * hf + r];
  __syncthreads();

  // write unnormalized exp(P) f16 into LDS [row i][j]
  #pragma unroll
  for (int st = 0; st < 8; ++st) {
    #pragma unroll
    for (int r = 0; r < 8; ++r)
      Pl[(8 * hf + r) * 2048 + wave * 128 + st * 16 + lane15] = (_Float16)S[st][r];
  }
  __syncthreads();

  // ---- Phase 2: out tile = P x V^T ; wave owns c-chunk wave*16 ----
  int cw = wave * 16;
  const _Float16* Vrow = Pall + (size_t)(b * M320 + 64 + cw + lane15) * L_ + hf * 16;
  const _Float16* Prow = Pl + lane15 * 2048 + 16 * hf;
  v8f accO = {};
  #pragma unroll 2
  for (int jj = 0; jj < 2048; jj += 32) {
    // A operand: the layout's two contiguous 8-half runs -> 2x aligned ds_load_b128
    union { v8h h[2]; v16h v; } a;
    a.h[0] = *(const v8h*)(Prow + jj);
    a.h[1] = *(const v8h*)(Prow + jj + 8);
    // V rows of Pall are already [c][j] f16: lane=col c, contiguous K=j halves
    v16h bV = *(const v16h*)(Vrow + jj);
    accO = __builtin_amdgcn_wmma_f32_16x16x32_f16(false, a.v, false, bV, (short)0,
                                                  accO, false, false);
  }

  // out = gamma * (acc / rowsum) + x   (rows per lane are contiguous in i)
  float g = gamma[0];
  #pragma unroll
  for (int r = 0; r < 8; ++r) {
    int row = 8 * hf + r;
    float scale = g / rs[r];
    size_t idx = (size_t)(b * C_ + cw + lane15) * L_ + i0 + row;
    out[idx] = scale * accO[r] + x[idx];
  }
}

// ---------------------------------------------------------------------------
extern "C" void kernel_launch(void* const* d_in, const int* in_sizes, int n_in,
                              void* d_out, int out_size, void* d_ws, size_t ws_size,
                              hipStream_t stream) {
  const float* x     = (const float*)d_in[0];
  const float* wq    = (const float*)d_in[1];
  const float* bq    = (const float*)d_in[2];
  const float* wk    = (const float*)d_in[3];
  const float* bk    = (const float*)d_in[4];
  const float* wv    = (const float*)d_in[5];
  const float* bv    = (const float*)d_in[6];
  const float* gamma = (const float*)d_in[7];
  float* out = (float*)d_out;

  char* ws = (char*)d_ws;
  _Float16* Wh   = (_Float16*)(ws + 0);          // 163,840 B  (A-swizzled weights)
  _Float16* Pall = (_Float16*)(ws + 163840);     // 10,485,760 B ([B][320][L] f16: Q|K|V)
  _Float16* Qh   = (_Float16*)(ws + 10649600);   // 1,048,576 B (A-swizzled Q tiles)
  _Float16* Kh   = (_Float16*)(ws + 11698176);   // 1,048,576 B (K^T [b][j][o])

  prepack_w<<<320, 256, 0, stream>>>(wq, wk, wv, Wh);
  proj_gemm<<<128, 512, 0, stream>>>(x, Wh, bq, bk, bv, Pall);
  repack_q<<<2048, 256, 0, stream>>>(Pall, Qh);
  repack_k<<<2048, 256, 0, stream>>>(Pall, Kh);
  attn_fused<<<1024, 512, 0, stream>>>(x, Pall, Qh, Kh, gamma, out);
}
